// ViT_Attention_48713519071710
// MI455X (gfx1250) — matrix operand order repaired and checked by reading, not compiled, not run on hardware
//
#include <hip/hip_runtime.h>
#include <cstdint>

#define DEV_INLINE __device__ __attribute__((always_inline)) inline

typedef __attribute__((ext_vector_type(16))) __bf16 v16bf;
typedef __attribute__((ext_vector_type(8)))  __bf16 v8bf;
typedef __attribute__((ext_vector_type(8)))  float  v8f;

#define WMMA_BF16(A, B, C) \
  __builtin_amdgcn_wmma_f32_16x16x32_bf16(false, (A), false, (B), (short)0, (C), false, false)

constexpr int Bb = 8, Nn = 2048, Dd = 768, Hh = 12, HD = 64;
constexpr int MROWS = Bb * Nn;          // 16384
constexpr float SCALE = 0.125f;         // 64^-0.5

// ---------------------------------------------------------------------------
// Async global->LDS copies (gfx1250 GLOBAL_LOAD_ASYNC_TO_LDS_B128, ASYNCcnt),
// with a register-staged fallback if the builtin is unavailable.
// Builtin prototype (from hipcc diagnostic): (int4 AS1*, int4 AS3*, imm, imm).
// ---------------------------------------------------------------------------
#if defined(__HIP_DEVICE_COMPILE__) && defined(__gfx1250__) && \
    __has_builtin(__builtin_amdgcn_global_load_async_to_lds_b128)
#define USE_ASYNC_LDS 1
#else
#define USE_ASYNC_LDS 0
#endif

#if USE_ASYNC_LDS
#if __has_builtin(__builtin_amdgcn_s_wait_asynccnt)
#define WAIT_ASYNC() __builtin_amdgcn_s_wait_asynccnt(0)
#else
#define WAIT_ASYNC() asm volatile("s_wait_asynccnt 0" ::: "memory")
#endif
#else
#define WAIT_ASYNC() ((void)0)
#endif

DEV_INLINE void copy16_issue(const __bf16* g, __bf16* l) {
#if USE_ASYNC_LDS
  typedef int v4i __attribute__((vector_size(16)));
  typedef __attribute__((address_space(1))) v4i as1_v4i;
  typedef __attribute__((address_space(3))) v4i as3_v4i;
  __builtin_amdgcn_global_load_async_to_lds_b128(
      (as1_v4i*)(uintptr_t)g, (as3_v4i*)(unsigned)(uintptr_t)l, 0, 0);
#else
  *(v8bf*)l = *(const v8bf*)g;
#endif
}

// ---------------------------------------------------------------------------
// WMMA fragment loaders (bf16, from LDS; contiguous k = 16*(lane/16)+s for A and B).
// All loads are 16 contiguous halves per lane => 2x ds_load_b128.
// ---------------------------------------------------------------------------

// A fragment: 16x32 tile at src (row-major [m][k], `stride` halves per row).
DEV_INLINE v16bf load_frag_a(const __bf16* src, int stride, int lane) {
  const int m = lane & 15;
  const int g = lane >> 4;
  const __bf16* row = src + m * stride;
  v16bf a;
#pragma unroll
  for (int h = 0; h < 8; ++h) a[h] = row[g * 16 + h];
#pragma unroll
  for (int h = 0; h < 8; ++h) a[h + 8] = row[g * 16 + 8 + h];
  return a;
}

// B fragment from transposed storage: src is row-major [n][k] (B[k][n] = src[n][k]).
// lane: n = lane%16; half h -> k = 16*(lane/16)+h  => 16 contiguous halves.
DEV_INLINE v16bf load_frag_bt(const __bf16* src, int stride, int lane) {
  const int n = lane & 15;
  const int g = lane >> 4;
  const __bf16* row = src + n * stride + g * 16;
  v16bf b;
#pragma unroll
  for (int h = 0; h < 16; ++h) b[h] = row[h];
  return b;
}

// ---------------------------------------------------------------------------
// Conversion kernels
// ---------------------------------------------------------------------------
__global__ void cvt_f32_to_bf16(const float* __restrict__ in, __bf16* __restrict__ out, int n) {
  int i = blockIdx.x * 256 + threadIdx.x;
  if (i < n) out[i] = (__bf16)in[i];
}

// fp32 [R][C] row-major  ->  bf16 [C][R] (transposed), 32x32 tiles via LDS.
__global__ __launch_bounds__(256) void cvt_transpose_bf16(
    const float* __restrict__ in, __bf16* __restrict__ out, int R, int C) {
  __shared__ float tile[32][33];
  const int c0 = blockIdx.x * 32, r0 = blockIdx.y * 32;
  const int tx = threadIdx.x & 31, ty = threadIdx.x >> 5;
  for (int i = ty; i < 32; i += 8)
    tile[i][tx] = in[(size_t)(r0 + i) * C + c0 + tx];
  __syncthreads();
  for (int i = ty; i < 32; i += 8)
    out[(size_t)(c0 + i) * R + r0 + tx] = (__bf16)tile[tx][i];
}

// ---------------------------------------------------------------------------
// 128x128 GEMM core: C = A[128,768] * Bt[128,768]^T tile; double-buffered LDS.
// A row-major [m][k]; Bt row-major [n][k] (weights pre-transposed).
// 256 threads = 8 waves (4x2); each wave owns 32x64 = 2x4 fragments.
// ---------------------------------------------------------------------------
struct GemmSmem {
  __attribute__((aligned(16))) __bf16 a[2][128][40];
  __attribute__((aligned(16))) __bf16 b[2][128][40];
};

DEV_INLINE void gemm_fill(const __bf16* __restrict__ A, const __bf16* __restrict__ Bt,
                          int m0, int n0, int k0,
                          __bf16 (&a)[128][40], __bf16 (&b)[128][40], int tid) {
  const int c0 = tid, c1 = tid + 256;  // chunk = 8 halves; 512 chunks per matrix
  copy16_issue(A + (size_t)(m0 + (c0 >> 2)) * Dd + k0 + (c0 & 3) * 8, &a[c0 >> 2][(c0 & 3) * 8]);
  copy16_issue(A + (size_t)(m0 + (c1 >> 2)) * Dd + k0 + (c1 & 3) * 8, &a[c1 >> 2][(c1 & 3) * 8]);
  copy16_issue(Bt + (size_t)(n0 + (c0 >> 2)) * Dd + k0 + (c0 & 3) * 8, &b[c0 >> 2][(c0 & 3) * 8]);
  copy16_issue(Bt + (size_t)(n0 + (c1 >> 2)) * Dd + k0 + (c1 & 3) * 8, &b[c1 >> 2][(c1 & 3) * 8]);
}

DEV_INLINE void gemm_core(const __bf16* __restrict__ A, const __bf16* __restrict__ Bt,
                          int m0, int n0, int lane, int wm, int wn,
                          GemmSmem& sm, v8f acc[2][4]) {
  const int tid = threadIdx.x;
  gemm_fill(A, Bt, m0, n0, 0, sm.a[0], sm.b[0], tid);
  for (int kt = 0; kt < Dd / 32; ++kt) {
    WAIT_ASYNC();
    __syncthreads();
    if (kt + 1 < Dd / 32)
      gemm_fill(A, Bt, m0, n0, (kt + 1) * 32, sm.a[(kt + 1) & 1], sm.b[(kt + 1) & 1], tid);
    const int s = kt & 1;
    v16bf a0 = load_frag_a(&sm.a[s][wm * 32][0], 40, lane);
    v16bf a1 = load_frag_a(&sm.a[s][wm * 32 + 16][0], 40, lane);
#pragma unroll
    for (int fn = 0; fn < 4; ++fn) {
      v16bf bf = load_frag_bt(&sm.b[s][wn * 64 + fn * 16][0], 40, lane);
      acc[0][fn] = WMMA_BF16(a0, bf, acc[0][fn]);
      acc[1][fn] = WMMA_BF16(a1, bf, acc[1][fn]);
    }
  }
}

// ---------------------------------------------------------------------------
// Kernel 1: QKV projection. C = x @ W_qkv + b_qkv, scattered bf16 into
// q/k: [B, H, N, 64];  v: [B, H, 64, N] (d-major, for transposed V fragments).
// ---------------------------------------------------------------------------
__global__ __launch_bounds__(256) void qkv_gemm_kernel(
    const __bf16* __restrict__ A, const __bf16* __restrict__ Bt,
    const float* __restrict__ bias, __bf16* __restrict__ qo,
    __bf16* __restrict__ ko, __bf16* __restrict__ vo) {
  __shared__ GemmSmem sm;
  const int tid = threadIdx.x, lane = tid & 31, wave = tid >> 5;
  const int wm = wave >> 1, wn = wave & 1;
  const int m0 = blockIdx.x * 128, n0 = blockIdx.y * 128;
  const int g = lane >> 4;

  v8f acc[2][4] = {};
  gemm_core(A, Bt, m0, n0, lane, wm, wn, sm, acc);

#pragma unroll
  for (int fn = 0; fn < 4; ++fn) {
    int col = n0 + wn * 64 + fn * 16 + (lane & 15);   // 0..2303
    float bv = bias[col];
    int which = col / Dd;
    int dcol = col % Dd;
    int h = dcol / HD, hd = dcol % HD;
#pragma unroll
    for (int fm = 0; fm < 2; ++fm) {
#pragma unroll
      for (int r = 0; r < 8; ++r) {
        int row = m0 + wm * 32 + fm * 16 + g * 8 + r;  // b*2048 + n
        int bidx = row >> 11;
        int nrow = row & 2047;
        __bf16 val = (__bf16)(acc[fm][fn][r] + bv);
        if (which == 2) {
          vo[(((size_t)(bidx * Hh + h)) * HD + hd) * Nn + nrow] = val;
        } else {
          __bf16* dst = which ? ko : qo;
          dst[(((size_t)(bidx * Hh + h)) * Nn + nrow) * HD + hd] = val;
        }
      }
    }
  }
}

// ---------------------------------------------------------------------------
// Kernel 2: flash attention. One (b,h) x 128-query block per workgroup.
// K/V tiles double-buffered; V stored d-major so V fragments are contiguous.
// Output bf16 to [B, N, H, 64] (= [B*N, 768]) for the out-proj GEMM.
// ---------------------------------------------------------------------------
DEV_INLINE void attn_fill_kv(const __bf16* __restrict__ Kb, const __bf16* __restrict__ Vbt,
                             int k0r, __bf16 (&kl)[64][72], __bf16 (&vl)[64][72], int tid) {
  const int c0 = tid, c1 = tid + 256;  // 512 chunks of 8 halves per tile
  copy16_issue(Kb + (size_t)((c0 >> 3) + k0r) * HD + (c0 & 7) * 8, &kl[c0 >> 3][(c0 & 7) * 8]);
  copy16_issue(Kb + (size_t)((c1 >> 3) + k0r) * HD + (c1 & 7) * 8, &kl[c1 >> 3][(c1 & 7) * 8]);
  copy16_issue(Vbt + (size_t)(c0 >> 3) * Nn + k0r + (c0 & 7) * 8, &vl[c0 >> 3][(c0 & 7) * 8]);
  copy16_issue(Vbt + (size_t)(c1 >> 3) * Nn + k0r + (c1 & 7) * 8, &vl[c1 >> 3][(c1 & 7) * 8]);
}

__global__ __launch_bounds__(256) void attn_flash_kernel(
    const __bf16* __restrict__ Q, const __bf16* __restrict__ K,
    const __bf16* __restrict__ Vt, __bf16* __restrict__ O) {
  __shared__ __attribute__((aligned(16))) __bf16 q_lds[128][72];
  __shared__ __attribute__((aligned(16))) __bf16 k_lds[2][64][72];
  __shared__ __attribute__((aligned(16))) __bf16 v_lds[2][64][72];
  __shared__ __attribute__((aligned(16))) __bf16 p_lds[8][16][72];

  const int tid = threadIdx.x;
  const int wave = tid >> 5;
  const int lane = tid & 31;
  const int g = lane >> 4;
  const int bh = blockIdx.x;          // 0..95
  const int q0 = blockIdx.y * 128;

  const size_t base = (size_t)bh * Nn * HD;
  const __bf16* Qb = Q + base;
  const __bf16* Kb = K + base;
  const __bf16* Vbt = Vt + base;      // [64][2048] d-major

  // Stage the 128x64 Q block + first K/V tile.
#pragma unroll
  for (int i = 0; i < 4; ++i) {
    int c = tid + i * 256;
    copy16_issue(Qb + (size_t)(q0 + (c >> 3)) * HD + (c & 7) * 8, &q_lds[c >> 3][(c & 7) * 8]);
  }
  attn_fill_kv(Kb, Vbt, 0, k_lds[0], v_lds[0], tid);
  WAIT_ASYNC();
  __syncthreads();

  v16bf aq0 = load_frag_a(&q_lds[wave * 16][0], 72, lane);
  v16bf aq1 = load_frag_a(&q_lds[wave * 16][32], 72, lane);

  v8f o_acc[4] = {};
  float mrow[8], lrow[8];
#pragma unroll
  for (int r = 0; r < 8; ++r) { mrow[r] = -1e30f; lrow[r] = 0.0f; }

  for (int kt = 0; kt < Nn / 64; ++kt) {
    WAIT_ASYNC();
    __syncthreads();
    if (kt + 1 < Nn / 64)
      attn_fill_kv(Kb, Vbt, (kt + 1) * 64, k_lds[(kt + 1) & 1], v_lds[(kt + 1) & 1], tid);
    const int sb = kt & 1;

    // S = Q * K^T for this wave's 16 rows x 64 keys (C-layout fragments).
    v8f s[4];
#pragma unroll
    for (int nt = 0; nt < 4; ++nt) {
      v8f a = {};
      a = WMMA_BF16(aq0, load_frag_bt(&k_lds[sb][nt * 16][0], 72, lane), a);
      a = WMMA_BF16(aq1, load_frag_bt(&k_lds[sb][nt * 16][32], 72, lane), a);
      s[nt] = a;
    }
#pragma unroll
    for (int nt = 0; nt < 4; ++nt)
#pragma unroll
      for (int r = 0; r < 8; ++r) s[nt][r] = s[nt][r] * SCALE;

    // Online softmax. Lane slot r <-> row g*8+r; row spans 16 lanes of one half.
#pragma unroll
    for (int r = 0; r < 8; ++r) {
      float mx = fmaxf(fmaxf(s[0][r], s[1][r]), fmaxf(s[2][r], s[3][r]));
#pragma unroll
      for (int msk = 1; msk < 16; msk <<= 1)
        mx = fmaxf(mx, __shfl_xor(mx, msk, 32));
      float mold = mrow[r];
      float mn = fmaxf(mold, mx);
      mrow[r] = mn;
      float alpha = __expf(mold - mn);
      float rsum = 0.0f;
#pragma unroll
      for (int nt = 0; nt < 4; ++nt) {
        float p = __expf(s[nt][r] - mn);
        s[nt][r] = p;
        rsum += p;
      }
#pragma unroll
      for (int msk = 1; msk < 16; msk <<= 1)
        rsum += __shfl_xor(rsum, msk, 32);
      lrow[r] = lrow[r] * alpha + rsum;
#pragma unroll
      for (int dt = 0; dt < 4; ++dt) o_acc[dt][r] = o_acc[dt][r] * alpha;
    }

    // P: C-layout -> A-layout via this wave's private LDS tile.
#pragma unroll
    for (int nt = 0; nt < 4; ++nt)
#pragma unroll
      for (int r = 0; r < 8; ++r)
        p_lds[wave][g * 8 + r][nt * 16 + (lane & 15)] = (__bf16)s[nt][r];

    asm volatile("s_wait_dscnt 0" ::: "memory");

    v16bf pa0 = load_frag_a(&p_lds[wave][0][0], 72, lane);
    v16bf pa1 = load_frag_a(&p_lds[wave][0][32], 72, lane);
#pragma unroll
    for (int dt = 0; dt < 4; ++dt) {
      o_acc[dt] = WMMA_BF16(pa0, load_frag_bt(&v_lds[sb][dt * 16][0], 72, lane), o_acc[dt]);
      o_acc[dt] = WMMA_BF16(pa1, load_frag_bt(&v_lds[sb][dt * 16][32], 72, lane), o_acc[dt]);
    }
  }

  // Normalize and store to [B, N, H, 64].
  const int b = bh / Hh;
  const int h = bh % Hh;
#pragma unroll
  for (int r = 0; r < 8; ++r) {
    float inv = 1.0f / lrow[r];
    int qrow = q0 + wave * 16 + g * 8 + r;
    size_t rowbase = (size_t)(b * Nn + qrow) * Dd + h * HD;
#pragma unroll
    for (int dt = 0; dt < 4; ++dt) {
      int d = dt * 16 + (lane & 15);
      O[rowbase + d] = (__bf16)(o_acc[dt][r] * inv);
    }
  }
}

// ---------------------------------------------------------------------------
// Kernel 3: output projection. out = attn_out @ W_out + b_out (fp32 result).
// ---------------------------------------------------------------------------
__global__ __launch_bounds__(256) void out_gemm_kernel(
    const __bf16* __restrict__ A, const __bf16* __restrict__ Bt,
    const float* __restrict__ bias, float* __restrict__ out) {
  __shared__ GemmSmem sm;
  const int tid = threadIdx.x, lane = tid & 31, wave = tid >> 5;
  const int wm = wave >> 1, wn = wave & 1;
  const int m0 = blockIdx.x * 128, n0 = blockIdx.y * 128;
  const int g = lane >> 4;

  v8f acc[2][4] = {};
  gemm_core(A, Bt, m0, n0, lane, wm, wn, sm, acc);

#pragma unroll
  for (int fn = 0; fn < 4; ++fn) {
    int col = n0 + wn * 64 + fn * 16 + (lane & 15);
    float bv = bias[col];
#pragma unroll
    for (int fm = 0; fm < 2; ++fm) {
#pragma unroll
      for (int r = 0; r < 8; ++r) {
        int row = m0 + wm * 32 + fm * 16 + g * 8 + r;
        out[(size_t)row * Dd + col] = acc[fm][fn][r] + bv;
      }
    }
  }
}

// ---------------------------------------------------------------------------
// Host launcher
// ---------------------------------------------------------------------------
extern "C" void kernel_launch(void* const* d_in, const int* in_sizes, int n_in,
                              void* d_out, int out_size, void* d_ws, size_t ws_size,
                              hipStream_t stream) {
  const float* x    = (const float*)d_in[0];
  const float* Wqkv = (const float*)d_in[1];
  const float* bqkv = (const float*)d_in[2];
  const float* Wout = (const float*)d_in[3];
  const float* bout = (const float*)d_in[4];
  float* out = (float*)d_out;

  char* ws = (char*)d_ws;
  size_t off = 0;
  auto take = [&](size_t elems) -> __bf16* {
    __bf16* p = (__bf16*)(ws + off);
    off += ((elems * sizeof(__bf16)) + 255) & ~(size_t)255;
    return p;
  };
  __bf16* x_bf    = take((size_t)MROWS * Dd);         // 25.2 MB, [16384][768]
  __bf16* wqkv_t  = take((size_t)Dd * 3 * Dd);        //  3.5 MB, [2304][768]
  __bf16* wout_t  = take((size_t)Dd * Dd);            //  1.2 MB, [768][768]
  __bf16* q_bf    = take((size_t)Bb * Hh * Nn * HD);  // [B,H,N,64]
  __bf16* k_bf    = take((size_t)Bb * Hh * Nn * HD);  // [B,H,N,64]
  __bf16* v_bf    = take((size_t)Bb * Hh * Nn * HD);  // [B,H,64,N]
  __bf16* o_bf    = take((size_t)MROWS * Dd);         // [16384][768]

  {
    int n = MROWS * Dd;
    cvt_f32_to_bf16<<<(n + 255) / 256, 256, 0, stream>>>(x, x_bf, n);
  }
  cvt_transpose_bf16<<<dim3(3 * Dd / 32, Dd / 32), 256, 0, stream>>>(Wqkv, wqkv_t, Dd, 3 * Dd);
  cvt_transpose_bf16<<<dim3(Dd / 32, Dd / 32), 256, 0, stream>>>(Wout, wout_t, Dd, Dd);

  qkv_gemm_kernel<<<dim3(MROWS / 128, (3 * Dd) / 128), 256, 0, stream>>>(
      x_bf, wqkv_t, bqkv, q_bf, k_bf, v_bf);

  attn_flash_kernel<<<dim3(Bb * Hh, Nn / 128), 256, 0, stream>>>(
      q_bf, k_bf, v_bf, o_bf);

  out_gemm_kernel<<<dim3(MROWS / 128, Dd / 128), 256, 0, stream>>>(
      o_bf, wout_t, bout, out);
}
